// effConvKAN3D_9311489097883
// MI455X (gfx1250) — matrix operand order repaired and checked by reading, not compiled, not run on hardware
//
#include <hip/hip_runtime.h>
#include <hip/hip_bf16.h>
#include <math.h>

typedef __attribute__((ext_vector_type(16))) _Float16 v16h;
typedef __attribute__((ext_vector_type(8)))  _Float16 v8h;
typedef __attribute__((ext_vector_type(8)))  float    v8f;

#define IN_CH      8
#define OUT_CH     16
#define IN_FEAT    216          // 8 * 27
#define NBASIS     9            // silu + 8 spline bases
#define KTOT       1944         // 216 * 9
#define KCOL       1952         // padded column stride (32B aligned in f16)
#define L3         32768        // 32*32*32 spatial positions
#define ASTRIDE    296          // LDS row stride in halves (conflict-free, 16B aligned)
#define WAVES_PER_BLOCK 4
#define ROWS_PER_BLOCK  (WAVES_PER_BLOCK * 16)

// ---------------------------------------------------------------------------
// Kernel 1: pack [base_weight | spline_weight*scaler] into f16, column-major
// per output channel: Wf16[o*KCOL + k], k = i*9 + c (c==0 -> base, 1..8 -> spline)
// ---------------------------------------------------------------------------
__global__ void kan_pack_weights(const float* __restrict__ base_w,
                                 const float* __restrict__ spline_w,
                                 const float* __restrict__ scaler,
                                 _Float16* __restrict__ Wf16) {
    int idx = blockIdx.x * blockDim.x + threadIdx.x;
    if (idx >= OUT_CH * KCOL) return;
    int o = idx / KCOL;
    int k = idx - o * KCOL;
    float v = 0.0f;
    if (k < KTOT) {
        int i = k / NBASIS;
        int c = k - i * NBASIS;
        if (c == 0) {
            v = base_w[o * IN_FEAT + i];
        } else {
            v = spline_w[(o * IN_FEAT + i) * 8 + (c - 1)] * scaler[o * IN_FEAT + i];
        }
    }
    Wf16[idx] = (_Float16)v;
}

// ---------------------------------------------------------------------------
// Kernel 2: fused unfold + silu + cubic B-spline basis + WMMA GEMM
// Each wave: 16 output rows x 16 output channels, K loop over 1944 (padded).
// ---------------------------------------------------------------------------
__global__ __launch_bounds__(128)
void kan_main(const float* __restrict__ x,
              const _Float16* __restrict__ Wf16,
              float* __restrict__ out) {
    __shared__ _Float16 A_lds[WAVES_PER_BLOCK * 16 * ASTRIDE];

    const int lane = threadIdx.x & 31;
    const int wave = threadIdx.x >> 5;
    const int rowBase = blockIdx.x * ROWS_PER_BLOCK + wave * 16;
    _Float16* Aw = &A_lds[wave * 16 * ASTRIDE];

    const int  nlo = lane & 15;      // row (A) / column (B,D) within tile
    const bool hi  = lane >= 16;

    v8f acc = {};

    for (int chunk = 0; chunk < 7; ++chunk) {
        const int featBase = chunk * 32;
        const int nfeat    = (chunk < 6) ? 32 : 24;
        const int nsteps   = (chunk < 6) ? 9  : 7;   // K-steps of 32 (216 -> pad 224)

        // Previous iteration's A-fragment ds loads must have drained before refill.
        asm volatile("s_wait_dscnt 0" ::: "memory");

        if (chunk == 6) {
            // zero pad local K 216..223 for every row
            int row = lane & 15;
            int grp = lane >> 4;
            _Float16* z = &Aw[row * ASTRIDE + 216 + grp * 4];
            z[0] = z[1] = z[2] = z[3] = (_Float16)0.0f;
        }

        // ---- fill A tile: 16 rows x nfeat features, 9 f16 values each ----
        for (int t = 0; t < 16; ++t) {
            if (2 * t >= nfeat) break;                 // uniform across wave
            int p  = lane + 32 * t;
            int m  = p & 15;
            int il = p >> 4;                           // local feature index
            int iFeat = featBase + il;

            int mg = rowBase + m;
            int d = mg >> 10, h = (mg >> 5) & 31, w = mg & 31;
            int ci = iFeat / 27;
            int r  = iFeat - ci * 27;
            int kd = r / 9, kh = (r / 3) % 3, kw = r % 3;
            int dd = d + kd - 1, hh = h + kh - 1, ww = w + kw - 1;

            float xv = 0.0f;
            if ((unsigned)dd < 32u && (unsigned)hh < 32u && (unsigned)ww < 32u)
                xv = x[((ci * 32 + dd) * 32 + hh) * 32 + ww];

            // silu(x) = x * sigmoid(x)
            float silu = xv / (1.0f + __expf(-xv));

            // cubic B-spline, uniform knots g[t] = (t-3)*0.4 - 1, t = 0..11
            float b0[11];
            #pragma unroll
            for (int j = 0; j < 11; ++j) {
                float gl = 0.4f * (float)(j - 3) - 1.0f;
                b0[j] = (xv >= gl && xv < gl + 0.4f) ? 1.0f : 0.0f;
            }
            float b1[10];
            #pragma unroll
            for (int j = 0; j < 10; ++j) {
                float gl = 0.4f * (float)(j - 3) - 1.0f;
                b1[j] = (xv - gl) * 2.5f * b0[j] + ((gl + 0.8f) - xv) * 2.5f * b0[j + 1];
            }
            float b2[9];
            #pragma unroll
            for (int j = 0; j < 9; ++j) {
                float gl = 0.4f * (float)(j - 3) - 1.0f;
                b2[j] = (xv - gl) * 1.25f * b1[j] + ((gl + 1.2f) - xv) * 1.25f * b1[j + 1];
            }
            float b3[8];
            #pragma unroll
            for (int j = 0; j < 8; ++j) {
                float gl = 0.4f * (float)(j - 3) - 1.0f;
                b3[j] = (xv - gl) * (1.0f / 1.2f) * b2[j] +
                        ((gl + 1.6f) - xv) * (1.0f / 1.2f) * b2[j + 1];
            }

            _Float16* dst = &Aw[m * ASTRIDE + il * NBASIS];
            dst[0] = (_Float16)silu;
            #pragma unroll
            for (int c = 0; c < 8; ++c) dst[1 + c] = (_Float16)b3[c];
        }

        // make the wave's LDS tile visible to fragment reads
        asm volatile("s_wait_dscnt 0" ::: "memory");

        // ---- WMMA over this chunk ----
        const _Float16* arow = &Aw[nlo * ASTRIDE];
        for (int st = 0; st < nsteps; ++st) {
            int kk0  = st * 32;
            int off1 = kk0 + (hi ? 16 : 0);    // A layout: lo lanes K 0..15, hi lanes K 16..31 (matches B)
            v8h alo = *(const v8h*)(arow + off1);
            v8h ahi = *(const v8h*)(arow + off1 + 8);
            v16h a;
            #pragma unroll
            for (int e = 0; e < 8; ++e) { a[e] = alo[e]; a[8 + e] = ahi[e]; }

            int kb = chunk * 288 + kk0 + (hi ? 16 : 0);  // B layout: hi lanes K +16
            v16h b = *(const v16h*)(Wf16 + nlo * KCOL + kb);

            acc = __builtin_amdgcn_wmma_f32_16x16x32_f16(
                      false, a, false, b, (short)0, acc, false, false);
        }
    }

    // ---- write D: VGPR r -> M = r (+8 for hi lanes), N = lane&15 ----
    const int o  = nlo;
    const int mb = rowBase + (hi ? 8 : 0);
    #pragma unroll
    for (int r = 0; r < 8; ++r)
        out[o * L3 + mb + r] = acc[r];
}

// ---------------------------------------------------------------------------
extern "C" void kernel_launch(void* const* d_in, const int* in_sizes, int n_in,
                              void* d_out, int out_size, void* d_ws, size_t ws_size,
                              hipStream_t stream) {
    (void)in_sizes; (void)n_in; (void)out_size; (void)ws_size;
    const float* x        = (const float*)d_in[0];
    const float* base_w   = (const float*)d_in[1];
    const float* spline_w = (const float*)d_in[2];
    const float* scaler   = (const float*)d_in[3];
    float*       out      = (float*)d_out;
    _Float16*    Wf16     = (_Float16*)d_ws;   // needs OUT_CH*KCOL*2 = 62464 bytes

    int wElems = OUT_CH * KCOL;
    kan_pack_weights<<<(wElems + 255) / 256, 256, 0, stream>>>(base_w, spline_w, scaler, Wf16);

    kan_main<<<L3 / ROWS_PER_BLOCK, 128, 0, stream>>>(x, Wf16, out);
}
